// SimpleGemmaAttention_59493886984386
// MI455X (gfx1250) — matrix operand order repaired and checked by reading, not compiled, not run on hardware
//
#include <hip/hip_runtime.h>
#include <hip/hip_bf16.h>

// ---------------------------------------------------------------------------
// SimpleGemmaAttention on MI455X (gfx1250), bf16 WMMA path.
//   B=4, S=2048, HIDDEN=1152, N_HEADS=4, N_KV=1, HEAD_DIM=256
// Pipeline:
//   1) cast hidden -> bf16; transpose+cast w_qkv, w_o -> bf16 (B-fragment layout)
//   2) WMMA GEMM (32x64 wave tiles): qkv(f32) = hidden_bf @ w_qkvT_bf
//   3) fused RMSNorm + RoPE + scaling -> q_bf, k_bf, v_t(bf16, d-major)
//   4) flash attention, 1 wave32 per 16-row q tile, bf16 WMMA, fp32 softmax,
//      global_prefetch for the next k-tile
//   5) WMMA GEMM: out(f32) = attn_bf @ w_oT_bf
// ---------------------------------------------------------------------------

typedef __bf16 bf16;
typedef bf16  v16bf __attribute__((ext_vector_type(16)));
typedef bf16  v8bf  __attribute__((ext_vector_type(8)));
typedef float v8f   __attribute__((ext_vector_type(8)));

#define B_SZ     4
#define S_LEN    2048
#define HID      1152
#define NH       4
#define HD       256
#define QKV_N    1536   // (4+2)*256
#define ATT_N    1024   // 4*256
#define TOKENS   (B_SZ * S_LEN)   // 8192

__device__ __forceinline__ v8f zero8() {
  v8f z = {0.f, 0.f, 0.f, 0.f, 0.f, 0.f, 0.f, 0.f};
  return z;
}

__device__ __forceinline__ v8f wmma_bf16(v16bf a, v16bf b, v8f c) {
  // D = A(16x32 bf16) * B(32x16 bf16) + C(16x16 f32)
  return __builtin_amdgcn_wmma_f32_16x16x32_bf16(
      /*neg_a=*/false, a, /*neg_b=*/false, b,
      /*c_mod=*/(short)0, c, /*reuse_a=*/false, /*reuse_b=*/false);
}

// A-fragment (16x32, 16-bit): lane = half*16 + m holds row m,
// elems 0..15 = K[16*half .. 16*half+15] (contiguous).
// Both chunks are contiguous 16B loads from a row-major source.
__device__ __forceinline__ v16bf load_a_frag(const bf16* row_k0, int ls) {
  v8bf lo = *(const v8bf*)(row_k0 + 16 * ls);
  v8bf hi = *(const v8bf*)(row_k0 + 16 * ls + 8);
  v16bf r;
#pragma unroll
  for (int i = 0; i < 8; ++i) { r[i] = lo[i]; r[i + 8] = hi[i]; }
  return r;
}

// ---------------------------------------------------------------------------
// 1) casts / transposes
// ---------------------------------------------------------------------------
__global__ void k_cast_bf16(const float* __restrict__ src, bf16* __restrict__ dst, int n) {
  int i = blockIdx.x * blockDim.x + threadIdx.x;
  if (i < n) dst[i] = (bf16)src[i];
}

// src [rows][cols] f32 -> dst [cols][rows] bf16
__global__ void k_transpose_bf16(const float* __restrict__ src, bf16* __restrict__ dst,
                                 int rows, int cols) {
  int i = blockIdx.x * blockDim.x + threadIdx.x;
  if (i < rows * cols) {
    int r = i / cols, c = i % cols;
    dst[(size_t)c * rows + r] = (bf16)src[i];
  }
}

// ---------------------------------------------------------------------------
// 2/5) WMMA GEMM: C[M][N] f32 = A[M][K] bf16 @ Bt[N][K] bf16
//     One wave computes a 32x64 tile (2 m-tiles x 4 n-tiles, 8 accumulators):
//     8 WMMA per 12 B128 loads. blockDim = 64 (2 independent waves).
// ---------------------------------------------------------------------------
__global__ __launch_bounds__(64) void k_gemm_bf16(const bf16* __restrict__ A,
                                                  const bf16* __restrict__ Bt,
                                                  float* __restrict__ C,
                                                  int M, int N, int K) {
  int strip   = blockIdx.x * 2 + (threadIdx.x >> 5);
  int nstrips = N / 64;
  int mbase   = (strip / nstrips) * 32;
  int nbase   = (strip % nstrips) * 64;
  if (mbase >= M) return;

  int lane = threadIdx.x & 31;
  int m    = lane & 15;
  int ls   = lane >> 4;

  v8f acc0[4], acc1[4];
#pragma unroll
  for (int j = 0; j < 4; ++j) { acc0[j] = zero8(); acc1[j] = zero8(); }

  const bf16* arow0 = A + (size_t)(mbase + m) * K;
  const bf16* arow1 = arow0 + (size_t)16 * K;
  const bf16* brow[4];
#pragma unroll
  for (int j = 0; j < 4; ++j) brow[j] = Bt + (size_t)(nbase + j * 16 + m) * K;

  for (int kc = 0; kc < K / 32; ++kc) {
    v16bf a0 = load_a_frag(arow0 + kc * 32, ls);
    v16bf a1 = load_a_frag(arow1 + kc * 32, ls);
#pragma unroll
    for (int j = 0; j < 4; ++j) {
      // B-fragment (32x16): lane (ls*16+n) holds col n, K = kc*32+16*ls .. +15 (contiguous)
      v16bf b = *(const v16bf*)(brow[j] + kc * 32 + 16 * ls);
      acc0[j] = wmma_bf16(a0, b, acc0[j]);
      acc1[j] = wmma_bf16(a1, b, acc1[j]);
    }
  }
#pragma unroll
  for (int j = 0; j < 4; ++j)
#pragma unroll
    for (int r = 0; r < 8; ++r) {
      C[(size_t)(mbase + r + 8 * ls) * N + nbase + j * 16 + m]      = acc0[j][r];
      C[(size_t)(mbase + 16 + r + 8 * ls) * N + nbase + j * 16 + m] = acc1[j][r];
    }
}

// ---------------------------------------------------------------------------
// 3) per-token RMSNorm + RoPE. 256 threads = 1 token.
//    q_bf [(b*4+h)][s][256], k_bf [b][s][256], v_t [b][d=256][s] (d-major!)
// ---------------------------------------------------------------------------
__global__ __launch_bounds__(256) void k_postprocess(const float* __restrict__ qkv,
                                                     const float* __restrict__ fcos,
                                                     const float* __restrict__ fsin,
                                                     const float* __restrict__ qw,
                                                     const float* __restrict__ kw,
                                                     bf16* __restrict__ qb,
                                                     bf16* __restrict__ kb,
                                                     bf16* __restrict__ vt) {
  int token = blockIdx.x;            // b*S + s
  int b = token / S_LEN, s = token % S_LEN;
  int t = threadIdx.x;               // 0..255
  int lane = t & 31, wid = t >> 5;

  const float* row = qkv + (size_t)token * QKV_N;

  // V: straight cast, stored d-major for the PV B-fragments
  vt[((size_t)b * HD + t) * S_LEN + s] = (bf16)row[ATT_N + HD + t];

  float cv = fcos[(size_t)s * 128 + (t & 127)];
  float sv = fsin[(size_t)s * 128 + (t & 127)];

  __shared__ float sh[HD];
  __shared__ float part[8];

  for (int hh = 0; hh < NH + 1; ++hh) {     // 0..3 = q heads, 4 = k
    const float* x = (hh < NH) ? row + hh * HD : row + ATT_N;
    const float* w = (hh < NH) ? qw : kw;
    float v = x[t];
    float ss = v * v;
#pragma unroll
    for (int off = 16; off > 0; off >>= 1) ss += __shfl_xor(ss, off);
    if (lane == 0) part[wid] = ss;
    __syncthreads();
    float tot = 0.f;
#pragma unroll
    for (int i = 0; i < 8; ++i) tot += part[i];
    float inv = rsqrtf(tot * (1.0f / HD) + 1e-6f);
    float nv = v * inv * (1.0f + w[t]);
    sh[t] = nv;
    __syncthreads();
    float o;
    if (t < 128) o = sh[t] * cv - sh[t + 128] * sv;
    else         o = sh[t - 128] * sv + sh[t] * cv;
    __syncthreads();   // protect sh/part reuse
    if (hh < NH)
      qb[(((size_t)b * NH + hh) * S_LEN + s) * HD + t] = (bf16)(o * 0.0625f); // fold d^-0.5
    else
      kb[((size_t)b * S_LEN + s) * HD + t] = (bf16)o;
  }
}

// ---------------------------------------------------------------------------
// 4) flash attention: 1 wave32 / block, one 16-row q tile, 32-col k tiles.
// ---------------------------------------------------------------------------
__global__ __launch_bounds__(32) void k_flash(const bf16* __restrict__ qb,
                                              const bf16* __restrict__ kb,
                                              const bf16* __restrict__ vt,
                                              bf16* __restrict__ attn) {
  const int qtiles = S_LEN / 16;
  int tile = blockIdx.x;
  int qt = tile % qtiles;
  int bh = tile / qtiles;               // b*4 + h
  int h = bh % NH, b = bh / NH;
  int qbase = qt * 16;

  int lane = threadIdx.x;
  int m  = lane & 15;
  int ls = lane >> 4;

  const bf16* qrow  = qb + ((size_t)bh * S_LEN + qbase + m) * HD;
  const bf16* kbp   = kb + (size_t)b * S_LEN * HD;
  const bf16* vbp   = vt + (size_t)b * HD * S_LEN;

  v8f O[16];
#pragma unroll
  for (int i = 0; i < 16; ++i) O[i] = zero8();
  float mrun[8], lrun[8];
#pragma unroll
  for (int r = 0; r < 8; ++r) { mrun[r] = -1e30f; lrun[r] = 0.f; }

  __shared__ bf16 Pl[16 * 32];

  int nkt = (qbase + 16 + 31) / 32;
  for (int kt = 0; kt < nkt; ++kt) {
    int kbase = kt * 32;
    // prefetch the next k-tile's K rows while we compute on this one
    if (kt + 1 < nkt) {
      const bf16* nk = kbp + (size_t)(kbase + 32 + m) * HD + 16 * ls;
      __builtin_prefetch(nk, 0, 1);            // -> global_prefetch_b8
      __builtin_prefetch(nk + 16 * HD, 0, 1);
    }
    v8f s0 = zero8(), s1 = zero8();
    const bf16* krow0 = kbp + (size_t)(kbase + m) * HD;
    const bf16* krow1 = krow0 + 16 * HD;
#pragma unroll
    for (int c = 0; c < 8; ++c) {                 // 8 x K=32 chunks over HD=256
      v16bf a  = load_a_frag(qrow + c * 32, ls);
      v16bf b0 = *(const v16bf*)(krow0 + c * 32 + 16 * ls);
      v16bf b1 = *(const v16bf*)(krow1 + c * 32 + 16 * ls);
      s0 = wmma_bf16(a, b0, s0);
      s1 = wmma_bf16(a, b1, s1);
    }
    // online softmax over the 16x32 score tile (fp32)
#pragma unroll
    for (int r = 0; r < 8; ++r) {
      int row = qbase + r + 8 * ls;
      float e0 = (kbase + m      <= row) ? s0[r] : -1e30f;
      float e1 = (kbase + m + 16 <= row) ? s1[r] : -1e30f;
      float mx = fmaxf(e0, e1);
#pragma unroll
      for (int off = 8; off > 0; off >>= 1) mx = fmaxf(mx, __shfl_xor(mx, off));
      float mnew = fmaxf(mrun[r], mx);
      float p0 = __expf(e0 - mnew);
      float p1 = __expf(e1 - mnew);
      float sum = p0 + p1;
#pragma unroll
      for (int off = 8; off > 0; off >>= 1) sum += __shfl_xor(sum, off);
      float corr = __expf(mrun[r] - mnew);
      lrun[r] = lrun[r] * corr + sum;
      mrun[r] = mnew;
#pragma unroll
      for (int nc = 0; nc < 16; ++nc) O[nc][r] *= corr;
      Pl[(r + 8 * ls) * 32 + m]      = (bf16)p0;
      Pl[(r + 8 * ls) * 32 + m + 16] = (bf16)p1;
    }
    __syncthreads();  // single-wave WG: lowers to S_NOP; enforces DS ordering
    v16bf pa = load_a_frag(&Pl[m * 32], ls);      // P as 16x32 A-fragment
    __syncthreads();
#pragma unroll
    for (int nc = 0; nc < 16; ++nc) {
      // V B-fragment: lane col = output d within chunk, rows = s (contiguous in v_t)
      const bf16* vrow = vbp + (size_t)(nc * 16 + m) * S_LEN + kbase + 16 * ls;
      v16bf bv = *(const v16bf*)vrow;
      O[nc] = wmma_bf16(pa, bv, O[nc]);
    }
  }
  // epilogue: normalize, store bf16 row-major [token][h*256+d]
#pragma unroll
  for (int r = 0; r < 8; ++r) {
    float invl = 1.0f / lrun[r];
    int row = qbase + r + 8 * ls;
    size_t base = ((size_t)b * S_LEN + row) * ATT_N + (size_t)h * HD;
#pragma unroll
    for (int nc = 0; nc < 16; ++nc)
      attn[base + nc * 16 + m] = (bf16)(O[nc][r] * invl);
  }
}

// ---------------------------------------------------------------------------
extern "C" void kernel_launch(void* const* d_in, const int* in_sizes, int n_in,
                              void* d_out, int out_size, void* d_ws, size_t ws_size,
                              hipStream_t stream) {
  const float* hidden = (const float*)d_in[0];
  // d_in[1] = attention_mask (causal; applied analytically)
  const float* fcos   = (const float*)d_in[2];
  const float* fsin   = (const float*)d_in[3];
  const float* w_qkv  = (const float*)d_in[4];
  const float* q_nw   = (const float*)d_in[5];
  const float* k_nw   = (const float*)d_in[6];
  const float* w_o    = (const float*)d_in[7];

  char* ws = (char*)d_ws;
  size_t off = 0;
  bf16*  hid_bf  = (bf16*)(ws + off); off += (size_t)TOKENS * HID * 2;       // 18.9 MB
  bf16*  wqkvT   = (bf16*)(ws + off); off += (size_t)QKV_N * HID * 2;        //  3.5 MB
  bf16*  woT     = (bf16*)(ws + off); off += (size_t)HID * ATT_N * 2;        //  2.4 MB
  float* qkv     = (float*)(ws + off); off += (size_t)TOKENS * QKV_N * 4;    // 50.3 MB
  bf16*  q_bf    = (bf16*)(ws + off); off += (size_t)B_SZ * NH * S_LEN * HD * 2; // 16.8 MB
  bf16*  k_bf    = (bf16*)(ws + off); off += (size_t)B_SZ * S_LEN * HD * 2;  //  4.2 MB
  bf16*  v_t     = (bf16*)(ws + off); off += (size_t)B_SZ * HD * S_LEN * 2;  //  4.2 MB
  bf16*  attn_bf = (bf16*)(ws + off); off += (size_t)TOKENS * ATT_N * 2;     // 16.8 MB

  // 1) operand preparation
  {
    int n = TOKENS * HID;
    k_cast_bf16<<<(n + 255) / 256, 256, 0, stream>>>(hidden, hid_bf, n);
  }
  {
    int n = HID * QKV_N;
    k_transpose_bf16<<<(n + 255) / 256, 256, 0, stream>>>(w_qkv, wqkvT, HID, QKV_N);
  }
  {
    int n = ATT_N * HID;
    k_transpose_bf16<<<(n + 255) / 256, 256, 0, stream>>>(w_o, woT, ATT_N, HID);
  }

  // 2) QKV projection: M=8192/32, N=1536/64 -> 256*24 = 6144 strips
  k_gemm_bf16<<<6144 / 2, 64, 0, stream>>>(hid_bf, wqkvT, qkv, TOKENS, QKV_N, HID);

  // 3) RMSNorm + RoPE + layout
  k_postprocess<<<TOKENS, 256, 0, stream>>>(qkv, fcos, fsin, q_nw, k_nw, q_bf, k_bf, v_t);

  // 4) flash attention: B*H*(S/16) = 2048 single-wave blocks
  k_flash<<<B_SZ * NH * (S_LEN / 16), 32, 0, stream>>>(q_bf, k_bf, v_t, attn_bf);

  // 5) output projection: M=8192/32, N=1152/64 -> 256*18 = 4608 strips
  k_gemm_bf16<<<4608 / 2, 64, 0, stream>>>(attn_bf, woT, (float*)d_out, TOKENS, HID, ATT_N);
}
